// Attention_42700564857486
// MI455X (gfx1250) — compile-verified, matrix operand order checked by reading, not run on hardware
//
#include <hip/hip_runtime.h>
#include <hip/hip_bf16.h>

// ---------------------------------------------------------------------------
// MHA forward for MI455X (gfx1250, wave32, WMMA).
// - All matmuls via v_wmma_f32_16x16x32_f16 (f32 accumulate).
// - Weights pre-transposed once so every LDS staging copy is contiguous 16B.
// - Staging uses GLOBAL_LOAD_ASYNC_TO_LDS_B128 via inline asm (ASYNCcnt path).
// - Flash attention: S x S scores never touch HBM.
// ---------------------------------------------------------------------------

typedef _Float16 half_t;
typedef __attribute__((ext_vector_type(16))) _Float16 v16h;
typedef __attribute__((ext_vector_type(8)))  _Float16 v8h;
typedef __attribute__((ext_vector_type(4)))  _Float16 v4h;
typedef __attribute__((ext_vector_type(8)))  float    v8f;
typedef __attribute__((ext_vector_type(4)))  float    v4f;

#define BATCH 2
#define SEQ   2048
#define HEADS 16
#define HDIM  64
#define HID   1024          // HEADS * HDIM
#define MROWS (BATCH * SEQ) // 4096

#if defined(__gfx1250__)
#define HAS_ASYNC_LDS 1
#else
#define HAS_ASYNC_LDS 0
#endif

// Stage 16 bytes global -> LDS via the CDNA5 async-to-LDS path.
// LDS operand: low 32 bits of the generic shared address ARE the LDS byte
// offset (generic = aperture_hi32 | lds_offset on amdgcn).
static __device__ __forceinline__ void stage16(const half_t* g, half_t* l) {
#if HAS_ASYNC_LDS
  unsigned lofs = (unsigned)(size_t)l;
  unsigned long long ga = (unsigned long long)(size_t)g;
  asm volatile("global_load_async_to_lds_b128 %0, %1, off"
               :: "v"(lofs), "v"(ga)
               : "memory");
#else
  *(v8h*)l = *(const v8h*)g;
#endif
}
static __device__ __forceinline__ void stage_wait() {
#if HAS_ASYNC_LDS
  asm volatile("s_wait_asynccnt 0x0" ::: "memory");
#endif
}

static __device__ __forceinline__ v8f wmma_f16(v16h a, v16h b, v8f c) {
  return __builtin_amdgcn_wmma_f32_16x16x32_f16(
      false, a, false, b, (short)0, c, false, false);
}

static __device__ __forceinline__ v16h frag_cat(v8h lo, v8h hi) {
  return __builtin_shufflevector(lo, hi, 0,1,2,3,4,5,6,7,8,9,10,11,12,13,14,15);
}

// WMMA A/B operand fragment from a contiguous 32-half row.
// ISA layout: lane = idx + 16*kh holds K = kh*8 + j (VGPR 0-3) and
// K = 16 + kh*8 + j (VGPR 4-7).
static __device__ __forceinline__ v16h frag_row32(const half_t* row, int kh) {
  v8h lo = *(const v8h*)(row + kh * 8);
  v8h hi = *(const v8h*)(row + 16 + kh * 8);
  return frag_cat(lo, hi);
}

// ---------------------------------------------------------------------------
// f32 -> f16 plain conversion (activations)
// ---------------------------------------------------------------------------
__global__ void cvt_f32_f16_kernel(const float* __restrict__ in,
                                   half_t* __restrict__ out, int n4) {
  int i = blockIdx.x * blockDim.x + threadIdx.x;
  if (i >= n4) return;
  v4f x = *(const v4f*)(in + (size_t)i * 4);
  v4h o;
  o[0] = (half_t)x[0]; o[1] = (half_t)x[1];
  o[2] = (half_t)x[2]; o[3] = (half_t)x[3];
  *(v4h*)(out + (size_t)i * 4) = o;
}

// ---------------------------------------------------------------------------
// f32 -> f16 transpose: W[K][N] -> WT[N][K] (one-time; weights reused 32x).
// 32x32 LDS tile, padded to kill bank conflicts.
// ---------------------------------------------------------------------------
__global__ void cvt_transpose_kernel(const float* __restrict__ in,
                                     half_t* __restrict__ out) {
  __shared__ float tile[32][33];
  const int t  = threadIdx.x;             // 256 threads
  const int k0 = blockIdx.y * 32;
  const int n0 = blockIdx.x * 32;
  {
    int idx = t * 4;
    int row = idx >> 5, col = idx & 31;
    v4f x = *(const v4f*)(in + (size_t)(k0 + row) * HID + n0 + col);
    tile[row][col + 0] = x[0]; tile[row][col + 1] = x[1];
    tile[row][col + 2] = x[2]; tile[row][col + 3] = x[3];
  }
  __syncthreads();
  int n = t >> 3, kq = (t & 7) * 4;
  v4h o;
#pragma unroll
  for (int i = 0; i < 4; ++i) o[i] = (half_t)tile[kq + i][n];
  *(v4h*)(out + (size_t)(n0 + n) * HID + k0 + kq) = o;
}

// ---------------------------------------------------------------------------
// Tiled GEMM: C[M x 1024] = A[M x 1024](f16) * B(f16, given as BT[N][K]) + bias
// Block tile 128x128, 8 waves (4M x 2N), wave tile 32x64 = 8 WMMAs per K-step.
// A and BT tiles staged to LDS via contiguous async 16B copies.
// ---------------------------------------------------------------------------
__global__ void gemm_hh_kernel(const half_t* __restrict__ A,
                               const half_t* __restrict__ BT,
                               const float* __restrict__ bias,
                               half_t* __restrict__ outH,
                               float* __restrict__ outF) {
  __shared__ __align__(16) half_t ldsA[128 * 32];   // 8 KB
  __shared__ __align__(16) half_t ldsBT[128 * 32];  // 8 KB

  const int t     = threadIdx.x;       // 0..255 (8 waves)
  const int lane  = t & 31;
  const int w     = t >> 5;
  const int waveM = w & 3;             // 4 M-subtiles of 32 rows
  const int waveN = w >> 2;            // 2 N-subtiles of 64 cols
  const int m0    = blockIdx.y * 128;
  const int n0    = blockIdx.x * 128;

  const int mfrag = lane & 15;
  const int kh    = lane >> 4;
  const int hi    = lane >> 4;
  const int nl    = lane & 15;

  // Cooperative staging: 16 halves per thread per matrix.
  const int flat = t * 16;
  const int srow = flat >> 5;          // 0..127
  const int scol = flat & 31;          // 0 or 16

  v8f cc[2][4] = {};

  for (int k0 = 0; k0 < HID; k0 += 32) {
    __syncthreads();
    const half_t* ag = A  + (size_t)(m0 + srow) * HID + k0 + scol;
    const half_t* bg = BT + (size_t)(n0 + srow) * HID + k0 + scol;
    stage16(ag,     &ldsA[flat]);
    stage16(ag + 8, &ldsA[flat + 8]);
    stage16(bg,     &ldsBT[flat]);
    stage16(bg + 8, &ldsBT[flat + 8]);
    if (k0 + 32 < HID) {
      __builtin_prefetch(ag + 32, 0, 1);
      __builtin_prefetch(bg + 32, 0, 1);
    }
    stage_wait();
    __syncthreads();

    v16h af[2], bf[4];
#pragma unroll
    for (int i = 0; i < 2; ++i)
      af[i] = frag_row32(&ldsA[(waveM * 32 + i * 16 + mfrag) * 32], kh);
#pragma unroll
    for (int j = 0; j < 4; ++j)
      bf[j] = frag_row32(&ldsBT[(waveN * 64 + j * 16 + mfrag) * 32], kh);
#pragma unroll
    for (int i = 0; i < 2; ++i)
#pragma unroll
      for (int j = 0; j < 4; ++j)
        cc[i][j] = wmma_f16(af[i], bf[j], cc[i][j]);
  }

  // Epilogue. C layout: VGPR r -> (M = r + 8*hi, N = nl).
#pragma unroll
  for (int j = 0; j < 4; ++j) {
    const int col = n0 + waveN * 64 + j * 16 + nl;
    const float b = bias[col];
#pragma unroll
    for (int i = 0; i < 2; ++i) {
#pragma unroll
      for (int r = 0; r < 8; ++r) {
        const int row = m0 + waveM * 32 + i * 16 + r + 8 * hi;
        float val = cc[i][j][r] + b;
        if (outF) outF[(size_t)row * HID + col] = val;
        else      outH[(size_t)row * HID + col] = (half_t)val;
      }
    }
  }
}

// ---------------------------------------------------------------------------
// RoPE + layout change. src f16 [B,S,HID].
//   transpose_out == 0 : dst [B,H,S,D]   (q, k; rope applied, q pre-scaled)
//   transpose_out == 1 : dst [B,H,D,S]   (v; no rope) -> contiguous V^T rows
// ---------------------------------------------------------------------------
__global__ void rope_transpose_kernel(const half_t* __restrict__ src,
                                      const float* __restrict__ cosT,
                                      const float* __restrict__ sinT,
                                      half_t* __restrict__ dst,
                                      int apply_rope, int transpose_out,
                                      float scale) {
  int idx = blockIdx.x * blockDim.x + threadIdx.x;  // over B*S*H*(D/2)
  if (idx >= BATCH * SEQ * HEADS * (HDIM / 2)) return;
  int p = idx & 31;
  int r = idx >> 5;
  int h = r & (HEADS - 1);
  r >>= 4;
  int s = r & (SEQ - 1);
  int b = r >> 11;

  size_t sb = ((size_t)(b * SEQ + s)) * HID + h * HDIM + 2 * p;
  float e = (float)src[sb];
  float o = (float)src[sb + 1];
  float oe = e, oo = o;
  if (apply_rope) {
    float cc = cosT[s * 32 + p];
    float ss = sinT[s * 32 + p];
    oe = e * cc - o * ss;
    oo = e * ss + o * cc;
  }
  if (transpose_out) {
    size_t db = ((size_t)(b * HEADS + h) * HDIM + 2 * p) * SEQ + s;
    dst[db]       = (half_t)(oe * scale);
    dst[db + SEQ] = (half_t)(oo * scale);
  } else {
    size_t db = (((size_t)(b * HEADS + h) * SEQ) + s) * HDIM + 2 * p;
    dst[db]     = (half_t)(oe * scale);
    dst[db + 1] = (half_t)(oo * scale);
  }
}

// ---------------------------------------------------------------------------
// Flash attention (causal). One block = 4 waves = 64 query rows of one (b,h).
// KV in 32-wide chunks: K [32 x 64] row-major, V^T [64 x 32] — both staged
// with contiguous async 16B copies. 8 WMMAs per chunk per wave.
// ---------------------------------------------------------------------------
__global__ void flash_attn_kernel(const half_t* __restrict__ q,
                                  const half_t* __restrict__ k,
                                  const half_t* __restrict__ vT,
                                  half_t* __restrict__ o) {
  __shared__ __align__(16) half_t ldsK[32 * 64];
  __shared__ __align__(16) half_t ldsVT[64 * 32];
  __shared__ __align__(16) half_t ldsP[4][16 * 32];

  const int t    = threadIdx.x;   // 0..127 (4 waves)
  const int lane = t & 31;
  const int w    = t >> 5;
  const int nl   = lane & 15;
  const int hi   = lane >> 4;
  const int kh   = lane >> 4;

  const int gq = blockIdx.x & 31;       // S/64 query blocks
  const int bh = blockIdx.x >> 5;
  const int h  = bh & (HEADS - 1);
  const int b  = bh >> 4;
  const int q0 = gq * 64;
  const int q0w = q0 + w * 16;

  const size_t headBase  = ((size_t)(b * HEADS + h)) * SEQ * HDIM;
  const half_t* vThead   = vT + headBase;   // [D][S] rows

  const half_t* qrow = q + headBase + (size_t)(q0w + nl) * HDIM;
  v16h qa0 = frag_row32(qrow, kh);
  v16h qa1 = frag_row32(qrow + 32, kh);

  v8f acc[4] = {};
  float mrow[8], lrow[8];
#pragma unroll
  for (int r = 0; r < 8; ++r) { mrow[r] = -1e30f; lrow[r] = 0.f; }

  const int nchunks = (q0 >> 5) + 2;
  for (int cidx = 0; cidx < nchunks; ++cidx) {
    const int j0 = cidx * 32;
    __syncthreads();
    {
      const int flat = t * 16;
      // K tile: 32 kv rows x 64 features, row-major.
      const int krow = flat >> 6, kcol = flat & 63;
      const half_t* ks = k + headBase + (size_t)(j0 + krow) * HDIM + kcol;
      stage16(ks,     &ldsK[flat]);
      stage16(ks + 8, &ldsK[flat + 8]);
      // V^T tile: 64 feature rows x 32 kv, contiguous from [B,H,D,S].
      const int vrow = flat >> 5, vcol = flat & 31;
      const half_t* vs = vThead + (size_t)vrow * SEQ + j0 + vcol;
      stage16(vs,     &ldsVT[flat]);
      stage16(vs + 8, &ldsVT[flat + 8]);
      stage_wait();
    }
    __syncthreads();

    if (j0 <= q0w + 15) {
      // scores 16x32 = Q(16x64) . K^T
      v8f s_lo = {}, s_hi = {};
      s_lo = wmma_f16(qa0, frag_row32(&ldsK[nl * 64], kh), s_lo);
      s_lo = wmma_f16(qa1, frag_row32(&ldsK[nl * 64] + 32, kh), s_lo);
      s_hi = wmma_f16(qa0, frag_row32(&ldsK[(16 + nl) * 64], kh), s_hi);
      s_hi = wmma_f16(qa1, frag_row32(&ldsK[(16 + nl) * 64] + 32, kh), s_hi);

      // online softmax (rows r + 8*hi); half-wave shuffle reductions
#pragma unroll
      for (int r = 0; r < 8; ++r) {
        const int row = q0w + r + 8 * hi;
        float slo = ((j0 + nl)      <= row) ? s_lo[r] : -1e9f;
        float shi = ((j0 + 16 + nl) <= row) ? s_hi[r] : -1e9f;
        float pm = fmaxf(slo, shi);
        for (int off = 1; off < 16; off <<= 1)
          pm = fmaxf(pm, __shfl_xor(pm, off, 32));
        float mnew  = fmaxf(mrow[r], pm);
        float alpha = __expf(mrow[r] - mnew);
        float plo = __expf(slo - mnew);
        float phi = __expf(shi - mnew);
        float rs = plo + phi;
        for (int off = 1; off < 16; off <<= 1)
          rs += __shfl_xor(rs, off, 32);
        lrow[r] = lrow[r] * alpha + rs;
        mrow[r] = mnew;
#pragma unroll
        for (int tt = 0; tt < 4; ++tt) acc[tt][r] *= alpha;
        ldsP[w][(r + 8 * hi) * 32 + nl]      = (half_t)plo;
        ldsP[w][(r + 8 * hi) * 32 + 16 + nl] = (half_t)phi;
      }

      // O += P(16x32) . V(32x64)
      v16h pa = frag_row32(&ldsP[w][nl * 32], kh);
#pragma unroll
      for (int tt = 0; tt < 4; ++tt) {
        v16h vb = frag_row32(&ldsVT[(tt * 16 + nl) * 32], kh);
        acc[tt] = wmma_f16(pa, vb, acc[tt]);
      }
    }
  }

  // Normalize; write O in [B,S,H*D] layout (f16).
#pragma unroll
  for (int r = 0; r < 8; ++r) {
    const float inv = 1.0f / lrow[r];
    const int srow = q0w + r + 8 * hi;
    half_t* dst = o + ((size_t)(b * SEQ + srow)) * HID + h * HDIM;
#pragma unroll
    for (int tt = 0; tt < 4; ++tt)
      dst[tt * 16 + nl] = (half_t)(acc[tt][r] * inv);
  }
}

// ---------------------------------------------------------------------------
// Host-side orchestration
// ---------------------------------------------------------------------------
extern "C" void kernel_launch(void* const* d_in, const int* in_sizes, int n_in,
                              void* d_out, int out_size, void* d_ws, size_t ws_size,
                              hipStream_t stream) {
  const float* x    = (const float*)d_in[0];
  const float* Wq   = (const float*)d_in[1];
  const float* bq   = (const float*)d_in[2];
  const float* Wk   = (const float*)d_in[3];
  const float* bk   = (const float*)d_in[4];
  const float* Wv   = (const float*)d_in[5];
  const float* bv   = (const float*)d_in[6];
  const float* Wo   = (const float*)d_in[7];
  const float* bo   = (const float*)d_in[8];
  // d_in[9] (mask) unused: causal mask computed analytically.
  const float* fcos = (const float*)d_in[10];
  const float* fsin = (const float*)d_in[11];
  float* out = (float*)d_out;

  // Workspace layout (halves). Total 36M halves = 72 MB.
  half_t* ws = (half_t*)d_ws;
  const size_t M1 = 1024 * 1024;
  half_t* xh  = ws;                 // [B,S,HID]            4M
  half_t* wqh = ws + 4 * M1;        // W^T [N][K], 1M each
  half_t* wkh = ws + 5 * M1;
  half_t* wvh = ws + 6 * M1;
  half_t* woh = ws + 7 * M1;
  half_t* qt  = ws + 8 * M1;        // [B,S,HID] pre-rope
  half_t* kt  = ws + 12 * M1;
  half_t* vt  = ws + 16 * M1;
  half_t* qh  = ws + 20 * M1;       // [B,H,S,D]
  half_t* kh  = ws + 24 * M1;
  half_t* vh  = ws + 28 * M1;       // [B,H,D,S]  (V transposed)
  half_t* oh  = ws + 32 * M1;       // [B,S,HID] attention output

  // 1) convert x; transpose+convert the four weight matrices
  cvt_f32_f16_kernel<<<(MROWS * HID / 4 + 255) / 256, 256, 0, stream>>>(x, xh, MROWS * HID / 4);
  dim3 tgrid(HID / 32, HID / 32);
  cvt_transpose_kernel<<<tgrid, 256, 0, stream>>>(Wq, wqh);
  cvt_transpose_kernel<<<tgrid, 256, 0, stream>>>(Wk, wkh);
  cvt_transpose_kernel<<<tgrid, 256, 0, stream>>>(Wv, wvh);
  cvt_transpose_kernel<<<tgrid, 256, 0, stream>>>(Wo, woh);

  // 2) QKV projections
  dim3 ggrid(HID / 128, MROWS / 128);
  gemm_hh_kernel<<<ggrid, 256, 0, stream>>>(xh, wqh, bq, qt, nullptr);
  gemm_hh_kernel<<<ggrid, 256, 0, stream>>>(xh, wkh, bk, kt, nullptr);
  gemm_hh_kernel<<<ggrid, 256, 0, stream>>>(xh, wvh, bv, vt, nullptr);

  // 3) RoPE + layout: q,k -> [B,H,S,D] (q pre-scaled by 1/sqrt(D)); v -> [B,H,D,S]
  const int npairs = BATCH * SEQ * HEADS * (HDIM / 2);
  rope_transpose_kernel<<<(npairs + 255) / 256, 256, 0, stream>>>(qt, fcos, fsin, qh, 1, 0, 0.125f);
  rope_transpose_kernel<<<(npairs + 255) / 256, 256, 0, stream>>>(kt, fcos, fsin, kh, 1, 0, 1.0f);
  rope_transpose_kernel<<<(npairs + 255) / 256, 256, 0, stream>>>(vt, fcos, fsin, vh, 0, 1, 1.0f);

  // 4) flash attention -> oh [B,S,HID]
  flash_attn_kernel<<<BATCH * HEADS * (SEQ / 64), 128, 0, stream>>>(qh, kh, vh, oh);

  // 5) output projection (f32 epilogue into d_out)
  gemm_hh_kernel<<<ggrid, 256, 0, stream>>>(oh, woh, bo, nullptr, out);
}
